// VGGQFNv2_82188494176711
// MI455X (gfx1250) — 8-bit matrix operand order repaired and checked by reading, compiled, not run on hardware
//
#include <hip/hip_runtime.h>
#include <stdint.h>

typedef int v8i __attribute__((ext_vector_type(8)));

#define BATCH 512

__global__ void k_quant_input(const float* __restrict__ x,
                              const float* __restrict__ alpha_p,
                              int8_t* __restrict__ q) {
  size_t i = (size_t)blockIdx.x * blockDim.x + threadIdx.x;
  const float inva = 1.0f / alpha_p[0];
  int c = (int)(i & 3);
  size_t pos = i >> 2;
  size_t b = pos >> 10;
  size_t yx = pos & 1023;
  float v = 0.0f;
  if (c < 3) v = x[(b * 3 + (size_t)c) * 1024 + yx];
  float t = fminf(fmaxf(v * inva, -1.0f), 1.0f) * 127.0f;
  q[i] = (int8_t)(int)rintf(t);
}

__global__ void k_quant_w(const float* __restrict__ w,
                          const float* __restrict__ alpha_p,
                          int8_t* __restrict__ qw,
                          int IC, int cshift, int Kpad) {
  const int k = blockIdx.x * blockDim.x + threadIdx.x;
  const int oc = blockIdx.y;
  const int Cpad = 1 << cshift;
  const int t = k >> cshift;
  const int c = k & (Cpad - 1);
  float v = 0.0f;
  if (t < 9 && c < IC) v = w[((size_t)oc * IC + c) * 9 + t];
  const float inva = 1.0f / alpha_p[0];
  float qv = fminf(fmaxf(v * inva, -1.0f), 1.0f) * 7.0f;
  qw[(size_t)oc * Kpad + k] = (int8_t)(int)rintf(qv);
}

template <bool SMALLC, bool SIGNED_B>
__global__ void k_conv_wmma(const int8_t* __restrict__ act,
                            const int8_t* __restrict__ wq,
                            float* __restrict__ outf,
                            int Wlog, int cshift, int OC, int Kpad, int Keff) {
  __shared__ __align__(16) int8_t shA[64 * 64];

  const int tid = threadIdx.x;
  const int lane = tid & 31;
  const int wave = tid >> 5;
  const int l15 = lane & 15;
  const int h = lane >> 4;
  const int W = 1 << Wlog;
  const int Cpad = 1 << cshift;

  const int ptile = blockIdx.x * 8 + wave;
  const int ocbase = blockIdx.y * 64;

  const int n = ptile * 16 + l15;
  const int b = n >> (2 * Wlog);
  const int yx = n & ((1 << (2 * Wlog)) - 1);
  const int y = yx >> Wlog;
  const int x = yx & (W - 1);

  const int sr = tid >> 2;
  const int sq = (tid & 3) << 4;
  const int8_t* __restrict__ gsrc = wq + (size_t)(ocbase + sr) * Kpad + sq;
  const uint32_t ldst = (uint32_t)(uintptr_t)(&shA[sr * 64 + sq]);

  v8i acc[4];
#pragma unroll
  for (int j = 0; j < 4; ++j)
#pragma unroll
    for (int r = 0; r < 8; ++r) acc[j][r] = 0;

  for (int kt = 0; kt < Kpad; kt += 64) {
    __syncthreads();
    {
      const uint64_t ga = (uint64_t)(uintptr_t)(gsrc + kt);
      asm volatile("global_load_async_to_lds_b128 %0, %1, off"
                   :: "v"(ldst), "v"(ga)
                   : "memory");
      asm volatile("s_wait_asynccnt 0x0" ::: "memory");
    }
    __syncthreads();

    v8i B;
#pragma unroll
    for (int r = 0; r < 8; ++r) B[r] = 0;
    if (SMALLC) {
#pragma unroll
      for (int v = 0; v < 8; ++v) {
        const int kb = kt + ((v & 3) << 2) + ((v >> 2) << 5) + (h << 4);
        int val = 0;
        if (kb < Keff) {
          const int t = kb >> cshift;
          const int c = kb & (Cpad - 1);
          const int ty = y + (t / 3) - 1;
          const int tx = x + (t % 3) - 1;
          if ((unsigned)ty < (unsigned)W && (unsigned)tx < (unsigned)W) {
            const size_t aidx =
                ((((size_t)b << Wlog) + (size_t)ty) << Wlog) + (size_t)tx;
            val = *(const int*)(act + (aidx << cshift) + c);
          }
        }
        B[v] = val;
      }
    } else {
      const int t = kt >> cshift;
      const int c0 = kt & (Cpad - 1);
      const int ty = y + (t / 3) - 1;
      const int tx = x + (t % 3) - 1;
      if ((unsigned)ty < (unsigned)W && (unsigned)tx < (unsigned)W) {
        const size_t aidx =
            ((((size_t)b << Wlog) + (size_t)ty) << Wlog) + (size_t)tx;
        const int8_t* __restrict__ bp = act + (aidx << cshift) + c0;
#pragma unroll
        for (int v = 0; v < 8; ++v)
          B[v] = *(const int*)(bp + ((v & 3) << 2) + ((v >> 2) << 5) + (h << 4));
      }
    }

#pragma unroll
    for (int j = 0; j < 4; ++j) {
      const int8_t* __restrict__ arow = &shA[(j * 16 + l15) * 64 + (h << 4)];
      v8i A;
#pragma unroll
      for (int g = 0; g < 4; ++g) {
        const int2 d = *(const int2*)(arow + ((g & 1) << 3) + ((g >> 1) << 5));
        A[2 * g] = d.x;
        A[2 * g + 1] = d.y;
      }
      acc[j] = __builtin_amdgcn_wmma_i32_16x16x64_iu8(
           true, A,  SIGNED_B, B, acc[j],
           false,  false);
    }
  }

  float* __restrict__ obase = outf + (size_t)n * OC + ocbase + (h << 3);
#pragma unroll
  for (int j = 0; j < 4; ++j)
#pragma unroll
    for (int r = 0; r < 8; ++r) obase[j * 16 + r] = (float)acc[j][r];
}

__global__ void k_bn_stats(const float* __restrict__ xin,
                           const float* __restrict__ gamma,
                           const float* __restrict__ beta,
                           float* __restrict__ st, int C, size_t NP) {
  const int c = blockIdx.x;
  float s = 0.0f, sq = 0.0f;
  for (size_t p = threadIdx.x; p < NP; p += blockDim.x) {
    float v = xin[p * (size_t)C + c];
    s += v;
    sq += v * v;
  }
  __shared__ float sh[512];
  sh[threadIdx.x] = s;
  sh[256 + threadIdx.x] = sq;
  __syncthreads();
  for (int w = 128; w > 0; w >>= 1) {
    if ((int)threadIdx.x < w) {
      sh[threadIdx.x] += sh[threadIdx.x + w];
      sh[256 + threadIdx.x] += sh[256 + threadIdx.x + w];
    }
    __syncthreads();
  }
  if (threadIdx.x == 0) {
    float mean = sh[0] / (float)NP;
    float var = sh[256] / (float)NP - mean * mean;
    float sc = gamma[c] * rsqrtf(var + 1e-5f);
    st[2 * c] = sc;
    st[2 * c + 1] = beta[c] - mean * sc;
  }
}

__global__ void k_bn_relu(float* __restrict__ xio, const float* __restrict__ st,
                          int cmask, size_t total) {
  size_t i = (size_t)blockIdx.x * blockDim.x + threadIdx.x;
  if (i >= total) return;
  int c = (int)(i & (size_t)cmask);
  float v = fmaf(xio[i], st[2 * c], st[2 * c + 1]);
  xio[i] = fmaxf(v, 0.0f);
}

__global__ void k_quant_act(const float* __restrict__ xin,
                            const float* __restrict__ alpha_p,
                            int8_t* __restrict__ q,
                            int Hin, int ocshift, int holog, int pool) {
  size_t i = (size_t)blockIdx.x * blockDim.x + threadIdx.x;
  const int C = 1 << ocshift;
  const int Ho = 1 << holog;
  const int c = (int)(i & (size_t)(C - 1));
  size_t p = i >> ocshift;
  const int xo = (int)(p & (size_t)(Ho - 1));
  const int yo = (int)((p >> holog) & (size_t)(Ho - 1));
  const int b = (int)(p >> (2 * holog));
  float v;
  if (pool) {
    const size_t base = (((size_t)b * Hin + 2 * yo) * Hin + 2 * xo) * C + c;
    const size_t rs = (size_t)Hin * C;
    float v00 = xin[base];
    float v01 = xin[base + C];
    float v10 = xin[base + rs];
    float v11 = xin[base + rs + C];
    v = fmaxf(fmaxf(v00, v01), fmaxf(v10, v11));
  } else {
    v = xin[(((size_t)b * Hin + yo) * Hin + xo) * C + c];
  }
  const float inva = 1.0f / alpha_p[0];
  float t = fminf(fmaxf(v * inva, 0.0f), 1.0f) * 15.0f;
  q[i] = (int8_t)(int)rintf(t);
}

__global__ void k_quant_fcw(const float* __restrict__ fw,
                            const float* __restrict__ alpha_p,
                            int8_t* __restrict__ qw) {
  const int k = blockIdx.x * blockDim.x + threadIdx.x;
  const int cls = blockIdx.y;
  const int s = k >> 9;
  const int c = k & 511;
  float v = 0.0f;
  if (cls < 10) v = fw[(size_t)cls * 8192 + c * 16 + s];
  const float inva = 1.0f / alpha_p[0];
  float qv = fminf(fmaxf(v * inva, -1.0f), 1.0f) * 7.0f;
  qw[(size_t)cls * 8192 + k] = (int8_t)(int)rintf(qv);
}

__global__ void k_fc_wmma(const int8_t* __restrict__ qa,
                          const int8_t* __restrict__ qw,
                          const float* __restrict__ bias,
                          const float* __restrict__ aalpha,
                          const float* __restrict__ walpha,
                          float* __restrict__ out) {
  const int lane = threadIdx.x & 31;
  const int wave = threadIdx.x >> 5;
  const int l15 = lane & 15;
  const int h = lane >> 4;
  const int mt = blockIdx.x * 8 + wave;

  v8i acc = {0, 0, 0, 0, 0, 0, 0, 0};
  const int8_t* __restrict__ arow = qa + (size_t)(mt * 16 + l15) * 8192;
  const int8_t* __restrict__ brow = qw + (size_t)l15 * 8192;

  for (int kt = 0; kt < 8192; kt += 64) {
    __builtin_prefetch(arow + kt + 256, 0, 0);
    v8i A, B;
#pragma unroll
    for (int v = 0; v < 8; ++v) {
      const int ka = kt + ((v & 3) << 2) + ((v >> 2) << 5) + (h << 4);
      A[v] = *(const int*)(arow + ka);
      const int kb = kt + ((v & 3) << 2) + ((v >> 2) << 5) + (h << 4);
      B[v] = *(const int*)(brow + kb);
    }
    acc = __builtin_amdgcn_wmma_i32_16x16x64_iu8(
         false, A,  true, B, acc, false, false);
  }

  if (l15 < 10) {
    const float scale = aalpha[0] * walpha[0] * (1.0f / 105.0f);
    const float bb = bias[l15];
#pragma unroll
    for (int r = 0; r < 8; ++r) {
      const int row = mt * 16 + r + (h << 3);
      out[(size_t)row * 10 + l15] = fmaf((float)acc[r], scale, bb);
    }
  }
}

extern "C" void kernel_launch(void* const* d_in, const int* in_sizes, int n_in,
                              void* d_out, int out_size, void* d_ws,
                              size_t ws_size, hipStream_t stream) {
  (void)in_sizes; (void)n_in; (void)out_size; (void)ws_size;
  const float* x = (const float*)d_in[0];
  auto P = [&](int i) { return (const float*)d_in[i]; };

  char* ws = (char*)d_ws;
  size_t off = 0;
  auto carve = [&](size_t bytes) -> char* {
    char* p = ws + off;
    off += (bytes + 255) & ~(size_t)255;
    return p;
  };
  float* convf = (float*)carve((size_t)BATCH * 32 * 32 * 128 * sizeof(float));
  int8_t* qA = (int8_t*)carve((size_t)BATCH * 32 * 32 * 128);
  int8_t* qB = (int8_t*)carve((size_t)BATCH * 32 * 32 * 128);
  int8_t* qw = (int8_t*)carve((size_t)512 * 4608);
  int8_t* qwfc = (int8_t*)carve((size_t)16 * 8192);
  float* st = (float*)carve(1024 * sizeof(float));

  k_quant_input<<<8192, 256, 0, stream>>>(x, P(4), qA);

  struct LI { int IC, cshift, Wlog, OC, pool; };
  const LI L[6] = {{3, 2, 5, 128, 0},  {128, 7, 5, 128, 1},
                   {128, 7, 4, 256, 0}, {256, 8, 4, 256, 1},
                   {256, 8, 3, 512, 0}, {512, 9, 3, 512, 1}};

  for (int li = 0; li < 6; ++li) {
    const int base = 1 + 6 * li;
    const LI& l = L[li];
    const int Cpad = 1 << l.cshift;
    const int Kpad = (li == 0) ? 64 : 9 * Cpad;
    const int Keff = (li == 0) ? 36 : Kpad;
    const int H = 1 << l.Wlog;
    const int NP = BATCH * H * H;
    const int8_t* qin = (li & 1) ? qB : qA;
    int8_t* qout = (li & 1) ? qA : qB;

    k_quant_w<<<dim3(Kpad / 64, l.OC), 64, 0, stream>>>(
        P(base + 0), P(base + 2), qw, l.IC, l.cshift, Kpad);

    dim3 cg(NP / 128, l.OC / 64);
    if (li == 0)
      k_conv_wmma<true, true><<<cg, 256, 0, stream>>>(
          qin, qw, convf, l.Wlog, l.cshift, l.OC, Kpad, Keff);
    else
      k_conv_wmma<false, false><<<cg, 256, 0, stream>>>(
          qin, qw, convf, l.Wlog, l.cshift, l.OC, Kpad, Keff);

    k_bn_stats<<<l.OC, 256, 0, stream>>>(convf, P(base + 4), P(base + 5), st,
                                         l.OC, (size_t)NP);
    const size_t total = (size_t)NP * l.OC;
    k_bn_relu<<<(unsigned)(total / 256), 256, 0, stream>>>(convf, st, l.OC - 1,
                                                           total);

    const float* nalpha = (li < 5) ? P(1 + 6 * (li + 1) + 3) : P(40);
    const int holog = l.pool ? (l.Wlog - 1) : l.Wlog;
    const int ocshift = (l.OC == 128) ? 7 : (l.OC == 256) ? 8 : 9;
    const size_t qtot = ((size_t)BATCH << (2 * holog)) << ocshift;
    k_quant_act<<<(unsigned)(qtot / 256), 256, 0, stream>>>(
        convf, nalpha, qout, H, ocshift, holog, l.pool);
  }

  k_quant_fcw<<<dim3(32, 16), 256, 0, stream>>>(P(37), P(39), qwfc);
  k_fc_wmma<<<4, 256, 0, stream>>>(qA, qwfc, P(38), P(40), P(39),
                                   (float*)d_out);
}
